// ResBlock_3350074491533
// MI455X (gfx1250) — matrix operand order repaired and checked by reading, not compiled, not run on hardware
//
#include <hip/hip_runtime.h>
#include <hip/hip_bf16.h>
#include <math.h>

// ---------------------------------------------------------------------------
// Types / WMMA helpers (CDNA5 gfx1250, wave32, v_wmma_f32_16x16x32_bf16)
// ---------------------------------------------------------------------------
typedef __bf16 bf16_t;
typedef __attribute__((ext_vector_type(16))) __bf16 v16bf;
typedef __attribute__((ext_vector_type(8)))  float  v8f;

union FragU { v16bf v; uint4 u[2]; };

// A-fragment, 16x32 bf16 (MxK), row-major source with leading dim `ld` (halfs).
// ISA 7.12.2: row M = lane&15; lane group kg = lane>>4 holds K runs
// [kg*8, +8) and [16+kg*8, +8) -> two contiguous 16B loads.
__device__ inline v16bf load_frag_a(const bf16_t* base, int ld, int m0, int k0, int lane) {
    int row = m0 + (lane & 15);
    int kg  = lane >> 4;
    const bf16_t* p = base + (size_t)row * ld + k0 + kg * 8;
    FragU f;
    f.u[0] = *(const uint4*)(p);        // K = k0+kg*8    .. +7
    f.u[1] = *(const uint4*)(p + 16);   // K = k0+16+kg*8 .. +7
    return f.v;
}

// B-fragment, 32x16 bf16 (KxN). Element (k, col) at base[col*ld + k]
// (each B column is a contiguous row of the source).
// Lane group kg = lane>>4 holds K runs [kg*8, +8) and [16+kg*8, +8), col = lane&15.
__device__ inline v16bf load_frag_b(const bf16_t* base, int ld, int col0, int k0, int lane) {
    int col = col0 + (lane & 15);
    int kg  = lane >> 4;
    const bf16_t* p = base + (size_t)col * ld + k0 + kg * 8;
    FragU f;
    f.u[0] = *(const uint4*)(p);        // K = k0+kg*8    .. +7
    f.u[1] = *(const uint4*)(p + 16);   // K = k0+16+kg*8 .. +7
    return f.v;
}

__device__ inline v8f wmma_bf16(v16bf a, v16bf b, v8f c) {
    return __builtin_amdgcn_wmma_f32_16x16x32_bf16(false, a, false, b, (short)0, c, false, false);
}

__device__ inline float redmax16(float v) {
    #pragma unroll
    for (int m = 1; m < 16; m <<= 1) v = fmaxf(v, __shfl_xor(v, m, 32));
    return v;
}
__device__ inline float redsum16(float v) {
    #pragma unroll
    for (int m = 1; m < 16; m <<= 1) v += __shfl_xor(v, m, 32);
    return v;
}

#define C_DIM 256
#define N_DIM 1024
#define B_DIM 64
#define EPS   1e-5f
#define W_LD  264   // padded LDS row stride (halfs) for 64x256 tiles: 528B = 33*16B
                    // -> B-frag lane start bank = (col*132 + kg*4) mod 64, distinct per lane

// Stage a 64x256 bf16 tile (row-major, ld_src halfs) into LDS with row stride W_LD.
// 128 threads, fully coalesced 16B segments.
__device__ inline void stage_tile_64x256(const bf16_t* src, int row0, int ld_src,
                                         bf16_t* dst, int tid) {
    for (int idx = tid; idx < 64 * 32; idx += 128) {   // 32 x 16B segments per row
        int r = idx >> 5, s = idx & 31;
        *(uint4*)(dst + r * W_LD + s * 8) =
            *(const uint4*)(src + (size_t)(row0 + r) * ld_src + s * 8);
    }
}

// ---------------------------------------------------------------------------
// Kernel 1: weight prep — fold BN into pointwise weights, convert to bf16.
// ---------------------------------------------------------------------------
__global__ void prep_kernel(const float* pw1, const float* pw2, const float* wv,
                            const float* wq, const float* wk,
                            const float* g1, const float* b1, const float* m1, const float* v1,
                            const float* g2, const float* b2, const float* m2, const float* v2,
                            bf16_t* Wp1, bf16_t* Wp2, bf16_t* Wvb,
                            bf16_t* Wqb, bf16_t* Wkb, float* bias1, float* bias2) {
    int i = blockIdx.x * 256 + threadIdx.x;      // 0 .. 65535
    int cout = i >> 8;
    float s1 = g1[cout] * rsqrtf(v1[cout] + EPS);
    float s2 = g2[cout] * rsqrtf(v2[cout] + EPS);
    Wp1[i] = (bf16_t)(pw1[i] * s1);
    Wp2[i] = (bf16_t)(pw2[i] * s2);
    Wvb[i] = (bf16_t)(wv[i]);
    if (i < 8192) { Wqb[i] = (bf16_t)wq[i]; Wkb[i] = (bf16_t)wk[i]; }
    if (i < 256) {
        float si1 = g1[i] * rsqrtf(v1[i] + EPS);
        float si2 = g2[i] * rsqrtf(v2[i] + EPS);
        bias1[i] = b1[i] - m1[i] * si1;
        bias2[i] = b2[i] - m2[i] * si2;
    }
}

// ---------------------------------------------------------------------------
// Kernel 2: depthwise 3x3 on NCHW fp32 input -> NHWC bf16 (LDS row cache).
// grid (H, B), 256 threads (one channel each).
// ---------------------------------------------------------------------------
__global__ void dw1_kernel(const float* __restrict__ x, const float* __restrict__ w9,
                           bf16_t* __restrict__ t0) {
    int h = blockIdx.x, b = blockIdx.y, tid = threadIdx.x;
    __shared__ bf16_t lds[3][C_DIM * 32];
    #pragma unroll
    for (int rr = 0; rr < 3; rr++) {
        int hr = h - 1 + rr;
        for (int idx = tid; idx < C_DIM * 32; idx += 256) {
            int c = idx >> 5, ww = idx & 31;
            float v = 0.f;
            if (hr >= 0 && hr < 32)
                v = x[((size_t)(b * C_DIM + c)) * N_DIM + hr * 32 + ww];
            lds[rr][idx] = (bf16_t)v;
        }
    }
    __syncthreads();
    int c = tid;
    float k[9];
    #pragma unroll
    for (int i = 0; i < 9; i++) k[i] = w9[c * 9 + i];
    for (int ww = 0; ww < 32; ww++) {
        float acc = 0.f;
        #pragma unroll
        for (int rr = 0; rr < 3; rr++)
            #pragma unroll
            for (int dc = 0; dc < 3; dc++) {
                int wn = ww + dc - 1;
                if (wn < 0 || wn > 31) continue;
                acc += (float)lds[rr][c * 32 + wn] * k[rr * 3 + dc];
            }
        t0[(((size_t)b * N_DIM) + h * 32 + ww) * C_DIM + c] = (bf16_t)acc;
    }
}

// ---------------------------------------------------------------------------
// Kernel 3: pointwise GEMM + folded BN (+ optional ReLU), NHWC in/out.
// Weight tile 64x256 staged once per block into LDS (shared by 4 waves).
// B fragments batched (4 live) so one dscnt wait covers a 4-WMMA burst.
// Y[n][c] = sum_k X[n][k] * W[c][k] + bias[c].  grid (64, B), 128 thr.
// ---------------------------------------------------------------------------
__global__ void gemm_bn_kernel(const bf16_t* __restrict__ X, const bf16_t* __restrict__ W,
                               const float* __restrict__ bias, bf16_t* __restrict__ Y,
                               int relu) {
    int b = blockIdx.y;
    int tn = blockIdx.x & 15, tc = blockIdx.x >> 4;
    int wave = threadIdx.x >> 5, lane = threadIdx.x & 31;
    int n0 = tn * 64 + wave * 16, c0 = tc * 64;
    const bf16_t* Xb = X + (size_t)b * N_DIM * C_DIM;

    __shared__ bf16_t ldsW[64 * W_LD];
    stage_tile_64x256(W, c0, C_DIM, ldsW, threadIdx.x);
    __syncthreads();

    v8f acc[4];
    #pragma unroll
    for (int j = 0; j < 4; j++) acc[j] = (v8f){0,0,0,0,0,0,0,0};
    for (int k0 = 0; k0 < C_DIM; k0 += 32) {
        v16bf a = load_frag_a(Xb, C_DIM, n0, k0, lane);
        v16bf bfr[4];
        #pragma unroll
        for (int j = 0; j < 4; j++) bfr[j] = load_frag_b(ldsW, W_LD, j * 16, k0, lane);
        #pragma unroll
        for (int j = 0; j < 4; j++) acc[j] = wmma_bf16(a, bfr[j], acc[j]);
    }
    int col_l = lane & 15, hi = lane >> 4;
    #pragma unroll
    for (int j = 0; j < 4; j++) {
        int c = c0 + j * 16 + col_l;
        float bs = bias[c];
        #pragma unroll
        for (int i = 0; i < 8; i++) {
            int m = n0 + i + 8 * hi;
            float v = acc[j][i] + bs;
            if (relu) v = fmaxf(v, 0.f);
            Y[((size_t)b * N_DIM + m) * C_DIM + c] = (bf16_t)v;
        }
    }
}

// ---------------------------------------------------------------------------
// Kernel 4: depthwise 3x3 directly on NHWC bf16 (fully coalesced taps).
// grid (N, B), 256 threads (one channel each).
// ---------------------------------------------------------------------------
__global__ void dw2_kernel(const bf16_t* __restrict__ a1, const float* __restrict__ w9,
                           bf16_t* __restrict__ t1) {
    int n = blockIdx.x, b = blockIdx.y, c = threadIdx.x;
    int h = n >> 5, ww = n & 31;
    float acc = 0.f;
    #pragma unroll
    for (int rr = 0; rr < 3; rr++) {
        int hr = h - 1 + rr;
        if (hr < 0 || hr > 31) continue;
        #pragma unroll
        for (int dc = 0; dc < 3; dc++) {
            int wn = ww + dc - 1;
            if (wn < 0 || wn > 31) continue;
            acc += (float)a1[((size_t)b * N_DIM + hr * 32 + wn) * C_DIM + c]
                 * w9[c * 9 + rr * 3 + dc];
        }
    }
    t1[((size_t)b * N_DIM + n) * C_DIM + c] = (bf16_t)acc;
}

// ---------------------------------------------------------------------------
// Kernel 5: pointwise GEMM #2 + BN + skip (a = y2 + x); LDS-staged weights,
// batched B fragments. Writes bf16 NHWC and fp32 [C][N].
// ---------------------------------------------------------------------------
__global__ void gemm2_fused_kernel(const bf16_t* __restrict__ X, const bf16_t* __restrict__ W,
                                   const float* __restrict__ bias, const float* __restrict__ xin,
                                   bf16_t* __restrict__ abf, float* __restrict__ af32) {
    int b = blockIdx.y;
    int tn = blockIdx.x & 15, tc = blockIdx.x >> 4;
    int wave = threadIdx.x >> 5, lane = threadIdx.x & 31;
    int n0 = tn * 64 + wave * 16, c0 = tc * 64;
    const bf16_t* Xb = X + (size_t)b * N_DIM * C_DIM;

    __shared__ bf16_t ldsW[64 * W_LD];
    stage_tile_64x256(W, c0, C_DIM, ldsW, threadIdx.x);
    __syncthreads();

    v8f acc[4];
    #pragma unroll
    for (int j = 0; j < 4; j++) acc[j] = (v8f){0,0,0,0,0,0,0,0};
    for (int k0 = 0; k0 < C_DIM; k0 += 32) {
        v16bf a = load_frag_a(Xb, C_DIM, n0, k0, lane);
        v16bf bfr[4];
        #pragma unroll
        for (int j = 0; j < 4; j++) bfr[j] = load_frag_b(ldsW, W_LD, j * 16, k0, lane);
        #pragma unroll
        for (int j = 0; j < 4; j++) acc[j] = wmma_bf16(a, bfr[j], acc[j]);
    }
    int col_l = lane & 15, hi = lane >> 4;
    #pragma unroll
    for (int j = 0; j < 4; j++) {
        int c = c0 + j * 16 + col_l;
        float bs = bias[c];
        #pragma unroll
        for (int i = 0; i < 8; i++) {
            int m = n0 + i + 8 * hi;
            size_t cn = ((size_t)b * C_DIM + c) * N_DIM + m;
            float v = acc[j][i] + bs + xin[cn];    // conv2 + BN2 + skip x
            af32[cn] = v;
            abf[((size_t)b * N_DIM + m) * C_DIM + c] = (bf16_t)v;
        }
    }
}

// ---------------------------------------------------------------------------
// Kernel 6: q/k projections. q[n][d] = sum_k a[n][k]*Wq[d][k] + bq[d]; d=32.
// grid (16, B), 128 thr; each wave: 16 rows, both q and k (2 tiles each).
// Weights are only 32KB total -> stay in L2/WGP$, no staging needed.
// ---------------------------------------------------------------------------
__global__ void qk_kernel(const bf16_t* __restrict__ abf,
                          const bf16_t* __restrict__ Wq, const bf16_t* __restrict__ Wk,
                          const float* __restrict__ bq, const float* __restrict__ bk,
                          bf16_t* __restrict__ q, bf16_t* __restrict__ k) {
    int b = blockIdx.y, nt = blockIdx.x;
    int wave = threadIdx.x >> 5, lane = threadIdx.x & 31;
    int n0 = nt * 64 + wave * 16;
    const bf16_t* Xb = abf + (size_t)b * N_DIM * C_DIM;
    v8f aq[2], ak[2];
    #pragma unroll
    for (int j = 0; j < 2; j++) { aq[j] = (v8f){0,0,0,0,0,0,0,0}; ak[j] = (v8f){0,0,0,0,0,0,0,0}; }
    for (int k0 = 0; k0 < C_DIM; k0 += 32) {
        v16bf a = load_frag_a(Xb, C_DIM, n0, k0, lane);
        v16bf bfr[4];
        #pragma unroll
        for (int j = 0; j < 2; j++) {
            bfr[j]     = load_frag_b(Wq, C_DIM, j * 16, k0, lane);
            bfr[2 + j] = load_frag_b(Wk, C_DIM, j * 16, k0, lane);
        }
        #pragma unroll
        for (int j = 0; j < 2; j++) {
            aq[j] = wmma_bf16(a, bfr[j],     aq[j]);
            ak[j] = wmma_bf16(a, bfr[2 + j], ak[j]);
        }
    }
    int col_l = lane & 15, hi = lane >> 4;
    #pragma unroll
    for (int j = 0; j < 2; j++) {
        int d = j * 16 + col_l;
        float bqv = bq[d], bkv = bk[d];
        #pragma unroll
        for (int i = 0; i < 8; i++) {
            int m = n0 + i + 8 * hi;
            size_t idx = ((size_t)b * N_DIM + m) * 32 + d;
            q[idx] = (bf16_t)(aq[j][i] + bqv);
            k[idx] = (bf16_t)(ak[j][i] + bkv);
        }
    }
}

// ---------------------------------------------------------------------------
// Kernel 7: v projection. v[c][n] = sum_k Wv[c][k]*a[n][k] + bv[c].
// The 64x256 activation tile (B operand, shared by all 4 waves) is LDS-staged;
// B fragments batched. Output layout [B][C][N]. grid (64, B); 128 thr.
// ---------------------------------------------------------------------------
__global__ void v_kernel(const bf16_t* __restrict__ Wv, const bf16_t* __restrict__ abf,
                         const float* __restrict__ bv, bf16_t* __restrict__ vout) {
    int b = blockIdx.y;
    int cg = blockIdx.x & 3, nt = blockIdx.x >> 2;
    int wave = threadIdx.x >> 5, lane = threadIdx.x & 31;
    int c0 = (cg * 4 + wave) * 16, ncol0 = nt * 64;
    const bf16_t* Xb = abf + (size_t)b * N_DIM * C_DIM;

    __shared__ bf16_t ldsX[64 * W_LD];
    stage_tile_64x256(Xb, ncol0, C_DIM, ldsX, threadIdx.x);
    __syncthreads();

    v8f acc[4];
    #pragma unroll
    for (int j = 0; j < 4; j++) acc[j] = (v8f){0,0,0,0,0,0,0,0};
    for (int k0 = 0; k0 < C_DIM; k0 += 32) {
        v16bf a = load_frag_a(Wv, C_DIM, c0, k0, lane);       // A = Wv rows (M=c)
        v16bf bfr[4];
        #pragma unroll
        for (int j = 0; j < 4; j++) bfr[j] = load_frag_b(ldsX, W_LD, j * 16, k0, lane);
        #pragma unroll
        for (int j = 0; j < 4; j++) acc[j] = wmma_bf16(a, bfr[j], acc[j]);
    }
    int col_l = lane & 15, hi = lane >> 4;
    #pragma unroll
    for (int j = 0; j < 4; j++) {
        #pragma unroll
        for (int i = 0; i < 8; i++) {
            int c = c0 + i + 8 * hi;
            int n = ncol0 + j * 16 + col_l;
            vout[((size_t)b * C_DIM + c) * N_DIM + n] = (bf16_t)(acc[j][i] + bv[c]);
        }
    }
}

// ---------------------------------------------------------------------------
// Kernel 8: flash-style attention + final residual/ReLU epilogue.
// Each wave owns one 16-row m-tile and the full 256-channel O accumulator
// (16 c-tiles x v8f). The 32-column V chunk [256][32] is LDS-staged once per
// block (all 4 waves consume it); P is reshaped f32-D -> bf16-A through padded
// LDS with an explicit s_wait_dscnt. V-fragments batched in groups of 4.
// grid (16, B), 128 thr (4 waves).
// ---------------------------------------------------------------------------
#define P_LD 40   // 80B stride -> conflict-free b128 LDS reads
#define V_LD 40   // same padding trick for the V chunk

__global__ void attn_kernel(const bf16_t* __restrict__ q, const bf16_t* __restrict__ k,
                            const bf16_t* __restrict__ v, const float* __restrict__ af32,
                            const float* __restrict__ gamma, float* __restrict__ out) {
    int b = blockIdx.y;
    int wave = threadIdx.x >> 5, lane = threadIdx.x & 31;
    int tid = threadIdx.x;
    int m0 = (blockIdx.x * 4 + wave) * 16;
    int col_l = lane & 15, hi = lane >> 4;

    const bf16_t* qb = q + (size_t)b * N_DIM * 32;
    const bf16_t* kb = k + (size_t)b * N_DIM * 32;
    const bf16_t* vb = v + (size_t)b * C_DIM * N_DIM;

    __shared__ bf16_t Pl[4][16 * P_LD];
    __shared__ bf16_t Vl[C_DIM * V_LD];
    bf16_t* myP = Pl[wave];

    v16bf aq = load_frag_a(qb, 32, m0, 0, lane);   // K = d = 32, one fragment

    float rowM[8], rowL[8];
    #pragma unroll
    for (int i = 0; i < 8; i++) { rowM[i] = -3.0e38f; rowL[i] = 0.f; }
    v8f o[16];
    #pragma unroll
    for (int ct = 0; ct < 16; ct++) o[ct] = (v8f){0,0,0,0,0,0,0,0};
    const v8f vzero = (v8f){0,0,0,0,0,0,0,0};

    for (int nb = 0; nb < N_DIM; nb += 32) {
        // Stage V chunk [256 c][32 n] -> LDS (16B segments, coalesced).
        for (int idx = tid; idx < C_DIM * 4; idx += 128) {
            int r = idx >> 2, s = idx & 3;
            *(uint4*)(Vl + r * V_LD + s * 8) =
                *(const uint4*)(vb + (size_t)r * N_DIM + nb + s * 8);
        }
        // Prefetch next chunk (global_prefetch_b8): 128 lanes x 2 rows = 256 rows.
        if (nb + 32 < N_DIM) {
            const bf16_t* nxt = vb + nb + 32;
            __builtin_prefetch(nxt + (size_t)(2 * tid) * N_DIM, 0, 3);
            __builtin_prefetch(nxt + (size_t)(2 * tid + 1) * N_DIM, 0, 3);
        }

        v16bf bk0 = load_frag_b(kb, 32, nb,      0, lane);
        v16bf bk1 = load_frag_b(kb, 32, nb + 16, 0, lane);
        v8f s0 = wmma_bf16(aq, bk0, vzero);
        v8f s1 = wmma_bf16(aq, bk1, vzero);

        #pragma unroll
        for (int i = 0; i < 8; i++) {
            float cmax = redmax16(fmaxf(s0[i], s1[i]));
            float newM = fmaxf(rowM[i], cmax);
            float corr = __expf(rowM[i] - newM);
            float p0 = __expf(s0[i] - newM);
            float p1 = __expf(s1[i] - newM);
            float rs = redsum16(p0 + p1);
            rowL[i] = rowL[i] * corr + rs;
            rowM[i] = newM;
            #pragma unroll
            for (int ct = 0; ct < 16; ct++) o[ct][i] *= corr;
            int m = i + 8 * hi;
            myP[m * P_LD + col_l]      = (bf16_t)p0;
            myP[m * P_LD + 16 + col_l] = (bf16_t)p1;
        }
        // own-wave P writes must land before the cross-lane A-fragment read
        asm volatile("s_wait_dscnt 0" ::: "memory");
        __syncthreads();   // V chunk staged & visible to all 4 waves

        v16bf ap = load_frag_a(myP, P_LD, 0, 0, lane);  // P as bf16 A-matrix
        #pragma unroll
        for (int ct0 = 0; ct0 < 16; ct0 += 4) {
            v16bf bvf[4];
            #pragma unroll
            for (int u = 0; u < 4; u++)
                bvf[u] = load_frag_b(Vl, V_LD, (ct0 + u) * 16, 0, lane);
            #pragma unroll
            for (int u = 0; u < 4; u++)
                o[ct0 + u] = wmma_bf16(ap, bvf[u], o[ct0 + u]);
        }
        __syncthreads();   // all waves done with Vl before next-chunk overwrite
    }

    float g = gamma[0];
    #pragma unroll
    for (int ct = 0; ct < 16; ct++) {
        int c = ct * 16 + col_l;
        #pragma unroll
        for (int i = 0; i < 8; i++) {
            int m = m0 + i + 8 * hi;
            size_t idx = ((size_t)b * C_DIM + c) * N_DIM + m;
            float ov = o[ct][i] / rowL[i];
            out[idx] = fmaxf(g * ov + af32[idx], 0.f);
        }
    }
}

// ---------------------------------------------------------------------------
// Host orchestration
// ---------------------------------------------------------------------------
extern "C" void kernel_launch(void* const* d_in, const int* in_sizes, int n_in,
                              void* d_out, int out_size, void* d_ws, size_t ws_size,
                              hipStream_t stream) {
    (void)in_sizes; (void)n_in; (void)out_size; (void)ws_size;

    const float* x    = (const float*)d_in[0];
    const float* dw1  = (const float*)d_in[1];
    const float* pw1  = (const float*)d_in[2];
    const float* g1   = (const float*)d_in[3];
    const float* b1   = (const float*)d_in[4];
    const float* m1   = (const float*)d_in[5];
    const float* v1   = (const float*)d_in[6];
    const float* dw2  = (const float*)d_in[7];
    const float* pw2  = (const float*)d_in[8];
    const float* g2   = (const float*)d_in[9];
    const float* b2   = (const float*)d_in[10];
    const float* m2   = (const float*)d_in[11];
    const float* v2   = (const float*)d_in[12];
    const float* wq   = (const float*)d_in[13];
    const float* bq   = (const float*)d_in[14];
    const float* wk   = (const float*)d_in[15];
    const float* bk   = (const float*)d_in[16];
    const float* wv   = (const float*)d_in[17];
    const float* bv   = (const float*)d_in[18];
    const float* gam  = (const float*)d_in[19];
    float* out = (float*)d_out;

    // Workspace layout (all 16B+ aligned)
    char* ws = (char*)d_ws;
    size_t off = 0;
    auto take = [&](size_t bytes) { char* p = ws + off; off += (bytes + 255) & ~(size_t)255; return p; };
    bf16_t* Wp1   = (bf16_t*)take(65536 * 2);
    bf16_t* Wp2   = (bf16_t*)take(65536 * 2);
    bf16_t* Wvb   = (bf16_t*)take(65536 * 2);
    bf16_t* Wqb   = (bf16_t*)take(8192 * 2);
    bf16_t* Wkb   = (bf16_t*)take(8192 * 2);
    float*  bias1 = (float*)take(256 * 4);
    float*  bias2 = (float*)take(256 * 4);
    size_t actH = (size_t)B_DIM * N_DIM * C_DIM;            // 16.7M elems
    bf16_t* t01 = (bf16_t*)take(actH * 2);                  // dw1 out, reused as dw2 out
    bf16_t* a1  = (bf16_t*)take(actH * 2);                  // dsconv1 out (NHWC)
    bf16_t* abf = (bf16_t*)take(actH * 2);                  // attention input (NHWC)
    float*  af32 = (float*)take(actH * 4);                  // attention input ([B][C][N] fp32)
    bf16_t* qb  = (bf16_t*)take((size_t)B_DIM * N_DIM * 32 * 2);
    bf16_t* kbuf = (bf16_t*)take((size_t)B_DIM * N_DIM * 32 * 2);
    bf16_t* vbuf = (bf16_t*)take(actH * 2);                 // [B][C][N]

    // 1. fold BN, convert weights to bf16
    prep_kernel<<<256, 256, 0, stream>>>(pw1, pw2, wv, wq, wk,
                                         g1, b1, m1, v1, g2, b2, m2, v2,
                                         Wp1, Wp2, Wvb, Wqb, Wkb, bias1, bias2);
    // 2. depthwise 1 (NCHW fp32 -> NHWC bf16)
    dw1_kernel<<<dim3(32, B_DIM), 256, 0, stream>>>(x, dw1, t01);
    // 3. pointwise 1 + BN1 + ReLU
    gemm_bn_kernel<<<dim3(64, B_DIM), 128, 0, stream>>>(t01, Wp1, bias1, a1, 1);
    // 4. depthwise 2 (NHWC)
    dw2_kernel<<<dim3(N_DIM, B_DIM), 256, 0, stream>>>(a1, dw2, t01);
    // 5. pointwise 2 + BN2 + skip x  -> abf (NHWC bf16) and af32 ([C][N] fp32)
    gemm2_fused_kernel<<<dim3(64, B_DIM), 128, 0, stream>>>(t01, Wp2, bias2, x, abf, af32);
    // 6. q/k projections
    qk_kernel<<<dim3(16, B_DIM), 128, 0, stream>>>(abf, Wqb, Wkb, bq, bk, qb, kbuf);
    // 7. v projection ([B][C][N])
    v_kernel<<<dim3(64, B_DIM), 128, 0, stream>>>(Wvb, abf, bv, vbuf);
    // 8. flash attention + gamma*O + a residual + final ReLU
    attn_kernel<<<dim3(16, B_DIM), 128, 0, stream>>>(qb, kbuf, vbuf, af32, gam, out);
}
